// GCNLayer_EdgeCat_20486994002066
// MI455X (gfx1250) — compile-verified, matrix operand order checked by reading, not run on hardware
//
#include <hip/hip_runtime.h>

// ---------------------------------------------------------------------------
// GCN layer with edge concat, fused: edge GEMM (bf16 WMMA) + mean segment
// reduce + node GEMM (bf16 WMMA), fp32 accumulate. Target: gfx1250 (CDNA5).
// ---------------------------------------------------------------------------

#define HID 128
#define EDGE_TILE 64
#define NODE_TILE 64
#define SA 392   // 384 + 8 bf16 pad  (row stride shifts 4 LDS banks/row)
#define SN 264   // 256 + 8 bf16 pad

typedef __attribute__((ext_vector_type(16))) __bf16        v16bf;
typedef __attribute__((ext_vector_type(8)))  float         v8f;
typedef __attribute__((ext_vector_type(8)))  unsigned int  v8u;

// round-to-nearest-even fp32 -> bf16
__device__ __forceinline__ unsigned short f2bf(float f) {
    union { float f; unsigned int u; } v; v.f = f;
    unsigned int r = v.u + 0x7FFFu + ((v.u >> 16) & 1u);
    return (unsigned short)(r >> 16);
}

// ISA 7.12.2: 16-bit A/B 16x32 K-mapping. Lane-half h, element e in [0,16):
//   K(e,h) = (e&7) + 8*h + 16*(e>>3)
// => elements 0..7 and 8..15 are each 8 contiguous K values (one b128 load).
__device__ __forceinline__ int kmap(int e, int h) {
    return (e & 7) + 8 * h + 16 * (e >> 3);
}

__device__ __forceinline__ v16bf load_frag_pair(const unsigned short* lo,
                                                const unsigned short* hi) {
    uint4 a0 = *(const uint4*)lo;
    uint4 a1 = *(const uint4*)hi;
    v8u u = {a0.x, a0.y, a0.z, a0.w, a1.x, a1.y, a1.z, a1.w};
    return __builtin_bit_cast(v16bf, u);
}

// ---------------------------------------------------------------------------
// Pre-swizzle weights into per-lane WMMA B-fragment order, bf16.
// Wbe layout: [ntile(8)][ktile(12)][lane(32)][elem(16)]  (49152 elems)
// Wbn layout: [ntile(8)][ktile(8)] [lane(32)][elem(16)]  (32768 elems)
// ---------------------------------------------------------------------------
__global__ __launch_bounds__(256) void prepack_kernel(
    const float* __restrict__ We, const float* __restrict__ Wn,
    unsigned short* __restrict__ Wbe, unsigned short* __restrict__ Wbn) {
    const int NE = 8 * 12 * 32 * 16;
    const int NN = 8 * 8 * 32 * 16;
    int i = blockIdx.x * 256 + threadIdx.x;
    if (i < NE) {
        int e  = i & 15;
        int l  = (i >> 4) & 31;
        int kt = (i >> 9) % 12;
        int nt = i / (512 * 12);
        int K   = kt * 32 + kmap(e, l >> 4);
        int col = nt * 16 + (l & 15);
        Wbe[i] = f2bf(We[(size_t)K * HID + col]);
    } else if (i < NE + NN) {
        int j  = i - NE;
        int e  = j & 15;
        int l  = (j >> 4) & 31;
        int kt = (j >> 9) & 7;
        int nt = j / (512 * 8);
        int K   = kt * 32 + kmap(e, l >> 4);
        int col = nt * 16 + (l & 15);
        Wbn[j] = f2bf(Wn[(size_t)K * HID + col]);
    }
}

// ---------------------------------------------------------------------------
// Edge kernel: m = ReLU(cat[nf[src], ef, nf[dst]] @ We + be)
//   out2 = m + efeats ;  hsum[dst] += m ;  cnt[dst] += 1
// Block = 256 threads (8 waves) handles 64 edges. Wave w owns N-tile w.
// ---------------------------------------------------------------------------
__global__ __launch_bounds__(256) void edge_kernel(
    const float* __restrict__ nfeats, const float* __restrict__ efeats,
    const int* __restrict__ src, const int* __restrict__ dst,
    const unsigned short* __restrict__ Wbe, const float* __restrict__ be,
    float* __restrict__ out2, float* __restrict__ hsum,
    float* __restrict__ cnt, int nEdges) {
    __shared__ unsigned short ldsA[EDGE_TILE * SA];
    __shared__ int sSrc[EDGE_TILE];
    __shared__ int sDst[EDGE_TILE];

    const int tid   = threadIdx.x;
    const int ebase = blockIdx.x * EDGE_TILE;

    if (tid < EDGE_TILE) {
        int eg = ebase + tid;
        int ec = eg < nEdges ? eg : nEdges - 1;
        sSrc[tid] = src[ec];
        sDst[tid] = dst[ec];
    }
    __syncthreads();

    if (tid < EDGE_TILE && (ebase + tid) < nEdges)
        unsafeAtomicAdd(&cnt[sDst[tid]], 1.0f);

    // Stage A: 64 edges x 384 cols, fp32 -> bf16, into padded LDS.
    for (int i = tid; i < EDGE_TILE * 96; i += 256) {
        int e   = i / 96;
        int col = (i % 96) * 4;
        int eg  = ebase + e;
        int ec  = eg < nEdges ? eg : nEdges - 1;
        const float* p;
        if (col < 128)      p = nfeats + (size_t)sSrc[e] * HID + col;
        else if (col < 256) p = efeats + (size_t)ec * HID + (col - 128);
        else                p = nfeats + (size_t)sDst[e] * HID + (col - 256);
        float4 f = *(const float4*)p;
        unsigned short* q = &ldsA[e * SA + col];
        q[0] = f2bf(f.x); q[1] = f2bf(f.y); q[2] = f2bf(f.z); q[3] = f2bf(f.w);
    }
    __syncthreads();

    const int w = tid >> 5, lane = tid & 31;
    const int n = lane & 15, h = lane >> 4;
    const int colbase = w * 16;

    float bval = be[colbase + n];
    v8f acc[4];
#pragma unroll
    for (int mt = 0; mt < 4; ++mt)
#pragma unroll
        for (int r = 0; r < 8; ++r) acc[mt][r] = bval;

    const unsigned short* wb_lane = Wbe + ((size_t)(w * 12) * 32 + lane) * 16;

#pragma unroll
    for (int kt = 0; kt < 12; ++kt) {
        const unsigned short* wp = wb_lane + (size_t)kt * 512;
        v16bf B = load_frag_pair(wp, wp + 8);
        const int kbase = kt * 32;
#pragma unroll
        for (int mt = 0; mt < 4; ++mt) {
            const unsigned short* ap = &ldsA[(mt * 16 + n) * SA + kbase + 8 * h];
            v16bf A = load_frag_pair(ap, ap + 16);
            acc[mt] = __builtin_amdgcn_wmma_f32_16x16x32_bf16(
                false, A, false, B, (short)0, acc[mt], false, false);
        }
    }

    // Epilogue: ReLU, fused residual store, atomic segment-sum.
#pragma unroll
    for (int mt = 0; mt < 4; ++mt) {
#pragma unroll
        for (int r = 0; r < 8; ++r) {
            int lrow = mt * 16 + r + 8 * h;   // C layout: M = r + 8*lane_half
            int eg   = ebase + lrow;
            if (eg < nEdges) {
                float v = fmaxf(acc[mt][r], 0.0f);
                size_t oi = (size_t)eg * HID + colbase + n;
                out2[oi] = v + efeats[oi];
                unsafeAtomicAdd(&hsum[(size_t)sDst[lrow] * HID + colbase + n], v);
            }
        }
    }
}

// ---------------------------------------------------------------------------
// Node kernel: h = hsum / max(cnt,1); out1 = ReLU(cat[nf,h] @ Wn + bn) + nf
// ---------------------------------------------------------------------------
__global__ __launch_bounds__(256) void node_kernel(
    const float* __restrict__ nfeats, const unsigned short* __restrict__ Wbn,
    const float* __restrict__ bn, const float* __restrict__ hsum,
    const float* __restrict__ cnt, float* __restrict__ out1, int nNodes) {
    __shared__ unsigned short ldsA[NODE_TILE * SN];
    __shared__ float sInv[NODE_TILE];

    const int tid   = threadIdx.x;
    const int nbase = blockIdx.x * NODE_TILE;

    if (tid < NODE_TILE) {
        int node = nbase + tid;
        float c  = (node < nNodes) ? cnt[node] : 1.0f;
        sInv[tid] = 1.0f / fmaxf(c, 1.0f);
    }
    __syncthreads();

    for (int i = tid; i < NODE_TILE * 64; i += 256) {
        int e    = i / 64;
        int col  = (i % 64) * 4;
        int node = nbase + e;
        float4 f = make_float4(0.f, 0.f, 0.f, 0.f);
        if (node < nNodes) {
            if (col < 128) {
                f = *(const float4*)(nfeats + (size_t)node * HID + col);
            } else {
                f = *(const float4*)(hsum + (size_t)node * HID + (col - 128));
                float inv = sInv[e];
                f.x *= inv; f.y *= inv; f.z *= inv; f.w *= inv;
            }
        }
        unsigned short* q = &ldsA[e * SN + col];
        q[0] = f2bf(f.x); q[1] = f2bf(f.y); q[2] = f2bf(f.z); q[3] = f2bf(f.w);
    }
    __syncthreads();

    const int w = tid >> 5, lane = tid & 31;
    const int n = lane & 15, h = lane >> 4;
    const int colbase = w * 16;

    float bval = bn[colbase + n];
    v8f acc[4];
#pragma unroll
    for (int mt = 0; mt < 4; ++mt)
#pragma unroll
        for (int r = 0; r < 8; ++r) acc[mt][r] = bval;

    const unsigned short* wb_lane = Wbn + ((size_t)(w * 8) * 32 + lane) * 16;

#pragma unroll
    for (int kt = 0; kt < 8; ++kt) {
        const unsigned short* wp = wb_lane + (size_t)kt * 512;
        v16bf B = load_frag_pair(wp, wp + 8);
        const int kbase = kt * 32;
#pragma unroll
        for (int mt = 0; mt < 4; ++mt) {
            const unsigned short* ap = &ldsA[(mt * 16 + n) * SN + kbase + 8 * h];
            v16bf A = load_frag_pair(ap, ap + 16);
            acc[mt] = __builtin_amdgcn_wmma_f32_16x16x32_bf16(
                false, A, false, B, (short)0, acc[mt], false, false);
        }
    }

#pragma unroll
    for (int mt = 0; mt < 4; ++mt) {
#pragma unroll
        for (int r = 0; r < 8; ++r) {
            int node = nbase + mt * 16 + r + 8 * h;
            if (node < nNodes) {
                size_t oi = (size_t)node * HID + colbase + n;
                out1[oi] = fmaxf(acc[mt][r], 0.0f) + nfeats[oi];
            }
        }
    }
}

// ---------------------------------------------------------------------------
// Host launcher.
// d_in: nfeats, efeats, src, dst, We, be, Wn, bn
// d_out: [N*H] (h_out + nfeats) then [E*H] (m + efeats)
// ws: Wbe(98304B) | Wbn(65536B) | hsum(N*H*4) | cnt(N*4)
// ---------------------------------------------------------------------------
extern "C" void kernel_launch(void* const* d_in, const int* in_sizes, int n_in,
                              void* d_out, int out_size, void* d_ws,
                              size_t ws_size, hipStream_t stream) {
    const float* nfeats = (const float*)d_in[0];
    const float* efeats = (const float*)d_in[1];
    const int*   src    = (const int*)d_in[2];
    const int*   dst    = (const int*)d_in[3];
    const float* We     = (const float*)d_in[4];
    const float* be     = (const float*)d_in[5];
    const float* Wn     = (const float*)d_in[6];
    const float* bn     = (const float*)d_in[7];

    const int E = in_sizes[2];
    const int N = in_sizes[0] / HID;

    char* ws = (char*)d_ws;
    unsigned short* Wbe = (unsigned short*)(ws);
    unsigned short* Wbn = (unsigned short*)(ws + 98304);
    float* hsum = (float*)(ws + 163840);
    float* cnt  = (float*)(ws + 163840 + (size_t)N * HID * sizeof(float));

    float* out1 = (float*)d_out;
    float* out2 = out1 + (size_t)N * HID;

    // zero accumulation buffers (graph-capturable async memset)
    hipMemsetAsync(ws + 163840, 0,
                   (size_t)N * HID * sizeof(float) + (size_t)N * sizeof(float),
                   stream);

    prepack_kernel<<<(8 * 12 * 512 + 8 * 8 * 512 + 255) / 256, 256, 0, stream>>>(
        We, Wn, Wbe, Wbn);

    edge_kernel<<<(E + EDGE_TILE - 1) / EDGE_TILE, 256, 0, stream>>>(
        nfeats, efeats, src, dst, Wbe, be, out2, hsum, cnt, E);

    node_kernel<<<(N + NODE_TILE - 1) / NODE_TILE, 256, 0, stream>>>(
        nfeats, Wbn, bn, hsum, cnt, out1, N);
}
